// ROPE_MultiheadAttention_24833500905948
// MI455X (gfx1250) — matrix operand order repaired and checked by reading, not compiled, not run on hardware
//
#include <hip/hip_runtime.h>
#include <hip/hip_bf16.h>

typedef __bf16 v16bf __attribute__((ext_vector_type(16)));
typedef float  v8f   __attribute__((ext_vector_type(8)));
typedef unsigned u32x4 __attribute__((ext_vector_type(4)));
typedef int      i32x4 __attribute__((ext_vector_type(4)));
typedef int      i32x8 __attribute__((ext_vector_type(8)));

#define DEV __device__ __forceinline__

static constexpr int BATCH = 2;
static constexpr int SEQ   = 2048;
static constexpr int DM    = 1024;
static constexpr int HEADS = 16;
static constexpr int DK    = 64;
static constexpr int MROWS = BATCH * SEQ;      // 4096
static constexpr int WAVES_PER_BLOCK = 8;      // 256 threads, wave32

DEV v8f wmma_bf16(v16bf a, v16bf b, v8f c) {
  return __builtin_amdgcn_wmma_f32_16x16x32_bf16(
      /*neg_a=*/false, a, /*neg_b=*/false, b,
      /*c_mod=*/(short)0, c, /*reuse_a=*/false, /*reuse_b=*/false);
}

// A-matrix (16x32 bf16) fragment: lane holds row m = row0 + lane%16.
// VGPR0..7 = K {base..base+15}, base = (lane/16)*16.
DEV v16bf load_a_frag(const __bf16* A, int lda, int row0, int k0, int lane) {
  const int m  = row0 + (lane & 15);
  const int kb = ((lane >> 4) & 1) * 16;
  const __bf16* p = A + (size_t)m * lda + k0;
  v16bf a;
#pragma unroll
  for (int e = 0; e < 8; ++e) a[e] = p[kb + e];
#pragma unroll
  for (int e = 0; e < 8; ++e) a[8 + e] = p[kb + 8 + e];
  return a;
}

// B-matrix (32x16 bf16) fragment from a ROW-MAJOR [N][K] source (i.e. B^T):
// B[k][n] = S[n][k]. Lane holds col n = col0 + lane%16; its 16 K values
// k = k0 + (lane/16)*16 + e are CONTIGUOUS -> one 32-byte load per lane.
DEV v16bf load_bT_frag(const __bf16* S, int lds_, int col0, int k0, int lane) {
  const int n  = col0 + (lane & 15);
  const int kb = ((lane >> 4) & 1) * 16;
  const __bf16* p = S + (size_t)n * lds_ + k0 + kb;
  v16bf b;
#pragma unroll
  for (int e = 0; e < 16; ++e) b[e] = p[e];
  return b;
}

// ---------------------------------------------------------------- TDM helper
// 2D tile load Global->LDS via Tensor Data Mover (D# per CDNA5 ISA ch.8).
// dim0   : tensor line length (elements, 2B each)
// dim1   : number of tensor lines
// tile0/1: tile size (elements / lines)
// stride0: elements between consecutive lines
DEV void tdm_load_2d(void* lds_ptr, const void* gptr,
                     unsigned dim0, unsigned dim1,
                     unsigned tile0, unsigned tile1, unsigned stride0) {
  const unsigned long long ga = (unsigned long long)gptr;
  const unsigned lds_addr = (unsigned)(size_t)lds_ptr;  // LDS offset (low 32 bits)
  u32x4 g0;
  g0[0] = 1u;                                           // count=1 (valid D#)
  g0[1] = lds_addr;                                     // lds_addr [63:32]
  g0[2] = (unsigned)ga;                                 // global_addr lo
  g0[3] = (unsigned)((ga >> 32) & 0x01FFFFFFull)        // global_addr[56:32]
        | 0x80000000u;                                  // type=2 ("image")
  i32x8 g1;
  g1[0] = 0x10000;                                      // data_size=1 (2 bytes)
  g1[1] = (int)((dim0 & 0xFFFFu) << 16);                // tensor_dim0 [79:48]
  g1[2] = (int)((dim0 >> 16) | ((dim1 & 0xFFFFu) << 16));
  g1[3] = (int)((dim1 >> 16) | (tile0 << 16));          // tile_dim0 [127:112]
  g1[4] = (int)tile1;                                   // tile_dim1; tile_dim2=0
  g1[5] = (int)stride0;                                 // tensor_dim0_stride lo
  g1[6] = 0;
  g1[7] = 0;
  const i32x4 z4 = {0, 0, 0, 0};                        // groups 2/3: unused (<=2D)
  const i32x8 z8 = {0, 0, 0, 0, 0, 0, 0, 0};
  __builtin_amdgcn_tensor_load_to_lds(g0, g1, z4, z4, z8, 0);
}

// ---------------------------------------------------------------- converters
__global__ void convert_f32_bf16(const float* __restrict__ src,
                                 __bf16* __restrict__ dst, int n) {
  int i = blockIdx.x * blockDim.x + threadIdx.x;
  if (i < n) dst[i] = (__bf16)src[i];
}

// dst[n*K + k] = src[k*N + n]  (bf16 transpose of a KxN f32 matrix)
__global__ void convert_transpose_f32_bf16(const float* __restrict__ src,
                                           __bf16* __restrict__ dst,
                                           int K, int N) {
  int i = blockIdx.x * blockDim.x + threadIdx.x;
  if (i < K * N) {
    int k = i % K;
    int n = i / K;
    dst[i] = (__bf16)src[(size_t)k * N + n];
  }
}

// ---------------------------------------------------------------- QKV + RoPE
// One wave computes a 16x64 tile (one head wide) of one of {Q,K,V}.
__global__ __launch_bounds__(256) void qkv_rope_kernel(
    const __bf16* __restrict__ xb,
    const __bf16* __restrict__ WqT, const __bf16* __restrict__ WkT,
    const __bf16* __restrict__ WvT, const int* __restrict__ tok,
    __bf16* __restrict__ Qo, __bf16* __restrict__ Ko, __bf16* __restrict__ Vt) {
  const int wave = threadIdx.x >> 5;
  const int lane = threadIdx.x & 31;
  const int wid  = blockIdx.x * WAVES_PER_BLOCK + wave;   // 0 .. 3*4096-1
  const int TPM  = (MROWS / 16) * (DM / 64);              // 4096 tiles / matrix
  const int w    = wid / TPM;
  const int rem  = wid % TPM;
  const int mt   = rem / (DM / 64);
  const int h    = rem % (DM / 64);                       // head == 64-col group
  const __bf16* WT = (w == 0) ? WqT : (w == 1) ? WkT : WvT;

  v8f c[4] = {v8f{}, v8f{}, v8f{}, v8f{}};
  const int arow = mt * 16 + (lane & 15);
  for (int kt = 0; kt < DM; kt += 32) {
    if (kt + 32 < DM) {
      __builtin_prefetch(xb + (size_t)arow * DM + kt + 32, 0, 3);
      __builtin_prefetch(WT + (size_t)(h * 64 + (lane & 15)) * DM + kt + 32, 0, 3);
    }
    v16bf a = load_a_frag(xb, DM, mt * 16, kt, lane);
#pragma unroll
    for (int t = 0; t < 4; ++t)
      c[t] = wmma_bf16(a, load_bT_frag(WT, DM, h * 64 + t * 16, kt, lane), c[t]);
  }

  const int hlf = (lane >> 4) & 1;
  if (w < 2) {
    __bf16* dst = (w == 0) ? Qo : Ko;
#pragma unroll
    for (int t = 0; t < 4; ++t) {
      const int dk = t * 16 + (lane & 15);
      const float expo = -((float)(dk & ~1)) / (float)DK;
      const float freq = __powf(10000.0f, expo);
      const bool  isEven = (dk & 1) == 0;
#pragma unroll
      for (int r = 0; r < 8; ++r) {
        const int m  = mt * 16 + hlf * 8 + r;
        const int bb = m / SEQ;
        const int s  = m % SEQ;
        const float pos = (float)tok[bb * SEQ + s];
        float sn, cs;
        __sincosf(pos * freq, &sn, &cs);
        const float self  = c[t][r];
        const float other = __shfl_xor(self, 1, 32);
        const float outv  = isEven ? (self * cs - other * sn)
                                   : (other * sn + self * cs);
        dst[(((size_t)bb * HEADS + h) * SEQ + s) * DK + dk] = (__bf16)outv;
      }
    }
  } else {
    // V stored TRANSPOSED: [B, H, DK, S] so PV B-fragments are contiguous.
#pragma unroll
    for (int t = 0; t < 4; ++t) {
      const int dk = t * 16 + (lane & 15);
#pragma unroll
      for (int r = 0; r < 8; ++r) {
        const int m  = mt * 16 + hlf * 8 + r;
        const int bb = m / SEQ;
        const int s  = m % SEQ;
        Vt[(((size_t)bb * HEADS + h) * DK + dk) * SEQ + s] = (__bf16)c[t][r];
      }
    }
  }
}

// ---------------------------------------------------------------- flash attention
// One wave per (b, h, 16-query tile). K/V 32-key tiles staged in LDS via TDM.
__global__ __launch_bounds__(256) void attn_kernel(
    const __bf16* __restrict__ Q, const __bf16* __restrict__ K,
    const __bf16* __restrict__ Vt, __bf16* __restrict__ O) {
  __shared__ __bf16 kbuf[WAVES_PER_BLOCK][32 * DK];   // [key][dk]   4 KB/wave
  __shared__ __bf16 vbuf[WAVES_PER_BLOCK][DK * 32];   // [dk][key]   4 KB/wave
  __shared__ __bf16 pbuf[WAVES_PER_BLOCK][16 * 32];   //             1 KB/wave

  const int wave = threadIdx.x >> 5;
  const int lane = threadIdx.x & 31;
  const int wid  = blockIdx.x * WAVES_PER_BLOCK + wave;   // 0..4095
  const int qt   = wid & (SEQ / 16 - 1);                  // 128 q-tiles
  const int bh   = wid >> 7;                              // b*HEADS + h
  const int b    = bh >> 4;
  const int h    = bh & 15;

  const __bf16* Qp = Q  + (size_t)bh * SEQ * DK;
  const __bf16* Kp = K  + (size_t)bh * SEQ * DK;
  const __bf16* Vh = Vt + (size_t)bh * DK * SEQ;          // [DK][SEQ]

  const int hlf = (lane >> 4) & 1;
  const int qrow_base = qt * 16 + hlf * 8;

  const v16bf qf0 = load_a_frag(Qp, DK, qt * 16, 0, lane);
  const v16bf qf1 = load_a_frag(Qp, DK, qt * 16, 32, lane);

  v8f acc0 = {}, acc1 = {}, acc2 = {}, acc3 = {};
  float mrow[8], lrow[8];
#pragma unroll
  for (int r = 0; r < 8; ++r) { mrow[r] = -1e30f; lrow[r] = 0.0f; }

  const __bf16* kl = &kbuf[wave][0];
  const __bf16* vl = &vbuf[wave][0];

  const int nkb = (qt * 16 + 15) / 32 + 1;   // causal 32-key blocks
  for (int kb = 0; kb < nkb; ++kb) {
    const int key0 = kb * 32;

    // Make sure previous block's LDS reads retired before TDM overwrites tiles.
    asm volatile("s_wait_dscnt 0" ::: "memory");
    // K tile: 32 keys x 64 dk from row-major [SEQ][DK].
    tdm_load_2d(&kbuf[wave][0], Kp + (size_t)key0 * DK,
                /*dim0=*/DK, /*dim1=*/SEQ - key0, /*tile0=*/DK, /*tile1=*/32,
                /*stride0=*/DK);
    // V tile: 64 dk-rows x 32 keys from transposed [DK][SEQ].
    tdm_load_2d(&vbuf[wave][0], Vh + key0,
                /*dim0=*/SEQ - key0, /*dim1=*/DK, /*tile0=*/32, /*tile1=*/DK,
                /*stride0=*/SEQ);
    __builtin_amdgcn_s_wait_tensorcnt(0);
    __builtin_amdgcn_wave_barrier();

    // scores from LDS K tile (local rows 0..31 = keys key0..key0+31)
    v8f s0 = {}, s1 = {};
    s0 = wmma_bf16(qf0, load_bT_frag(kl, DK, 0,  0,  lane), s0);
    s0 = wmma_bf16(qf1, load_bT_frag(kl, DK, 0,  32, lane), s0);
    s1 = wmma_bf16(qf0, load_bT_frag(kl, DK, 16, 0,  lane), s1);
    s1 = wmma_bf16(qf1, load_bT_frag(kl, DK, 16, 32, lane), s1);

    const int j0 = key0 + (lane & 15);
    const int j1 = j0 + 16;
    const float sc = 0.125f;   // 1/sqrt(64)

#pragma unroll
    for (int r = 0; r < 8; ++r) {
      const int q = qrow_base + r;
      float sv0 = (j0 <= q) ? s0[r] * sc : -1e30f;
      float sv1 = (j1 <= q) ? s1[r] * sc : -1e30f;

      float mx = fmaxf(sv0, sv1);
#pragma unroll
      for (int off = 1; off < 16; off <<= 1) mx = fmaxf(mx, __shfl_xor(mx, off, 32));
      const float mnew  = fmaxf(mrow[r], mx);
      const float alpha = __expf(mrow[r] - mnew);
      mrow[r] = mnew;

      const float p0 = __expf(sv0 - mnew);
      const float p1 = __expf(sv1 - mnew);
      float rs = p0 + p1;
#pragma unroll
      for (int off = 1; off < 16; off <<= 1) rs += __shfl_xor(rs, off, 32);
      lrow[r] = lrow[r] * alpha + rs;

      acc0[r] *= alpha; acc1[r] *= alpha; acc2[r] *= alpha; acc3[r] *= alpha;

      const int row = r + hlf * 8;
      pbuf[wave][row * 32 + (lane & 15)]      = (__bf16)p0;
      pbuf[wave][row * 32 + 16 + (lane & 15)] = (__bf16)p1;
    }

    // DS ops are in-order per wave; fence compiler + wait counter, then
    // re-read P in the A-matrix layout (16x32).
    asm volatile("s_wait_dscnt 0" ::: "memory");
    __builtin_amdgcn_wave_barrier();
    const v16bf pf = load_a_frag(&pbuf[wave][0], 32, 0, 0, lane);
    asm volatile("s_wait_dscnt 0" ::: "memory");
    __builtin_amdgcn_wave_barrier();

    // P(16x32) x V(32x64) from LDS V tile ([dk][key] layout).
    acc0 = wmma_bf16(pf, load_bT_frag(vl, 32, 0,  0, lane), acc0);
    acc1 = wmma_bf16(pf, load_bT_frag(vl, 32, 16, 0, lane), acc1);
    acc2 = wmma_bf16(pf, load_bT_frag(vl, 32, 32, 0, lane), acc2);
    acc3 = wmma_bf16(pf, load_bT_frag(vl, 32, 48, 0, lane), acc3);
  }

  // Normalize and store to [B, S, DM] (head-major columns) as bf16.
#pragma unroll
  for (int r = 0; r < 8; ++r) {
    const float inv = 1.0f / lrow[r];
    const int s = qrow_base + r;
    const size_t mbase = ((size_t)b * SEQ + s) * DM + h * DK;
    O[mbase +  0 + (lane & 15)] = (__bf16)(acc0[r] * inv);
    O[mbase + 16 + (lane & 15)] = (__bf16)(acc1[r] * inv);
    O[mbase + 32 + (lane & 15)] = (__bf16)(acc2[r] * inv);
    O[mbase + 48 + (lane & 15)] = (__bf16)(acc3[r] * inv);
  }
}

// ---------------------------------------------------------------- output projection
__global__ __launch_bounds__(256) void oproj_kernel(
    const __bf16* __restrict__ Ob, const __bf16* __restrict__ WoT,
    float* __restrict__ out) {
  const int wave = threadIdx.x >> 5;
  const int lane = threadIdx.x & 31;
  const int wid  = blockIdx.x * WAVES_PER_BLOCK + wave;  // 0 .. 4095
  const int mt   = wid / (DM / 64);
  const int ng   = wid % (DM / 64);

  v8f c[4] = {v8f{}, v8f{}, v8f{}, v8f{}};
  const int arow = mt * 16 + (lane & 15);
  for (int kt = 0; kt < DM; kt += 32) {
    if (kt + 32 < DM) {
      __builtin_prefetch(Ob + (size_t)arow * DM + kt + 32, 0, 3);
      __builtin_prefetch(WoT + (size_t)(ng * 64 + (lane & 15)) * DM + kt + 32, 0, 3);
    }
    v16bf a = load_a_frag(Ob, DM, mt * 16, kt, lane);
#pragma unroll
    for (int t = 0; t < 4; ++t)
      c[t] = wmma_bf16(a, load_bT_frag(WoT, DM, ng * 64 + t * 16, kt, lane), c[t]);
  }

  const int hlf = (lane >> 4) & 1;
#pragma unroll
  for (int t = 0; t < 4; ++t)
#pragma unroll
    for (int r = 0; r < 8; ++r) {
      const int m = mt * 16 + hlf * 8 + r;
      out[(size_t)m * DM + ng * 64 + t * 16 + (lane & 15)] = c[t][r];
    }
}

// ---------------------------------------------------------------- launch
extern "C" void kernel_launch(void* const* d_in, const int* in_sizes, int n_in,
                              void* d_out, int out_size, void* d_ws, size_t ws_size,
                              hipStream_t stream) {
  const float* x   = (const float*)d_in[0];
  const int*   tok = (const int*)d_in[1];
  const float* WQ  = (const float*)d_in[2];
  const float* WK  = (const float*)d_in[3];
  const float* WV  = (const float*)d_in[4];
  const float* WO  = (const float*)d_in[5];
  float* out = (float*)d_out;

  char* ws = (char*)d_ws;
  const size_t MiB = 1024 * 1024;
  __bf16* xb   = (__bf16*)(ws +  0 * MiB);  // [4096][1024] bf16      (8 MiB)
  __bf16* wqT  = (__bf16*)(ws +  8 * MiB);  // [1024][1024] bf16 (W^T, 2 MiB)
  __bf16* wkT  = (__bf16*)(ws + 10 * MiB);
  __bf16* wvT  = (__bf16*)(ws + 12 * MiB);
  __bf16* woT  = (__bf16*)(ws + 14 * MiB);
  __bf16* Qb   = (__bf16*)(ws + 16 * MiB);  // [B,H,S,DK]             (8 MiB)
  __bf16* Kb   = (__bf16*)(ws + 24 * MiB);
  __bf16* Vtb  = (__bf16*)(ws + 32 * MiB);  // [B,H,DK,S] (transposed)
  __bf16* Obuf = (__bf16*)(ws + 40 * MiB);  // [B,S,DM]               (8 MiB)

  const int NX = MROWS * DM;     // 4,194,304
  const int NW = DM * DM;        // 1,048,576
  convert_f32_bf16<<<(NX + 255) / 256, 256, 0, stream>>>(x, xb, NX);
  convert_transpose_f32_bf16<<<(NW + 255) / 256, 256, 0, stream>>>(WQ, wqT, DM, DM);
  convert_transpose_f32_bf16<<<(NW + 255) / 256, 256, 0, stream>>>(WK, wkT, DM, DM);
  convert_transpose_f32_bf16<<<(NW + 255) / 256, 256, 0, stream>>>(WV, wvT, DM, DM);
  convert_transpose_f32_bf16<<<(NW + 255) / 256, 256, 0, stream>>>(WO, woT, DM, DM);

  const int qkv_waves = 3 * (MROWS / 16) * (DM / 64);            // 12288
  qkv_rope_kernel<<<qkv_waves / WAVES_PER_BLOCK, 256, 0, stream>>>(
      xb, wqT, wkT, wvT, tok, Qb, Kb, Vtb);

  const int attn_waves = BATCH * HEADS * (SEQ / 16);             // 4096
  attn_kernel<<<attn_waves / WAVES_PER_BLOCK, 256, 0, stream>>>(Qb, Kb, Vtb, Obuf);

  const int op_waves = (MROWS / 16) * (DM / 64);                 // 4096
  oproj_kernel<<<op_waves / WAVES_PER_BLOCK, 256, 0, stream>>>(Obuf, woT, out);
}
